// PointSIS_Feature_Extractor_54674933678330
// MI455X (gfx1250) — matrix operand order repaired and checked by reading, not compiled, not run on hardware
//
#include <hip/hip_runtime.h>
#include <math.h>

// ---------------- CDNA5 WMMA types (probe-confirmed) ----------------
typedef __bf16 bf16;
typedef __attribute__((ext_vector_type(16))) __bf16 v16bf;
typedef __attribute__((ext_vector_type(8)))  float  v8f;
typedef __attribute__((ext_vector_type(4)))  unsigned v4u;
typedef __attribute__((ext_vector_type(8)))  int    v8i;
typedef __attribute__((ext_vector_type(4)))  int    v4i;

#if defined(__has_builtin)
#if __has_builtin(__builtin_amdgcn_tensor_load_to_lds)
#define HAVE_TDM 1
#endif
#endif
#ifndef HAVE_TDM
#define HAVE_TDM 0
#endif

static __device__ __forceinline__ bf16 tb(float f) {
    unsigned u = __builtin_bit_cast(unsigned, f);
    return __builtin_bit_cast(bf16, (unsigned short)(u >> 16));
}
static __device__ __forceinline__ int nrows(int rs, const int* rd) { return rd ? *rd : rs; }
static __device__ __forceinline__ float geluf(float x) { return 0.5f * x * (1.f + erff(x * 0.70710678118654752f)); }
static __device__ __forceinline__ float siluf(float x) { return x / (1.f + __expf(-x)); }
static __device__ __forceinline__ float softplusf(float x) { return x > 20.f ? x : log1pf(__expf(x)); }
static __device__ __forceinline__ float actf(float x, int a) { return a == 1 ? geluf(x) : (a == 2 ? softplusf(x) : x); }

#define PK4(v, o, q) { v[(o)] = tb((q).x); v[(o)+1] = tb((q).y); v[(o)+2] = tb((q).z); v[(o)+3] = tb((q).w); }

static __device__ __forceinline__ unsigned ldsoff(const void* p) {
    return (unsigned)(unsigned long long)p;    // generic shared ptr: low 32 bits = LDS offset
}

#if HAVE_TDM
// 2D TDM load: tile (tile1 rows x tile0 cols, 4B elements) from gaddr (row stride s0 elems) to LDS.
// flags_d0: data_size=4B + LDS pad configuration (ISA 8.4 group1 dword0).
static __device__ __forceinline__ void tdm_load_2d(const void* gaddr, unsigned lds_addr, unsigned flags_d0,
                                                   unsigned tdim0, unsigned tdim1,
                                                   unsigned tile0, unsigned tile1,
                                                   unsigned long long stride0)
{
    unsigned long long ga = (unsigned long long)gaddr;
    v4u g0;
    g0[0] = 1u;                                               // count=1, user descriptor
    g0[1] = lds_addr;                                         // lds_addr (bytes)
    g0[2] = (unsigned)ga;                                     // global_addr[31:0]
    g0[3] = (unsigned)((ga >> 32) & 0x01FFFFFFu) | (2u << 30);// global_addr[56:32] | type=2
    v8i g1;
    g1[0] = (int)flags_d0;                                    // wg_mask=0 | data_size | pad cfg
    g1[1] = (int)((tdim0 & 0xFFFFu) << 16);                   // tensor_dim0 lo16 (atomic addr 0)
    g1[2] = (int)((tdim0 >> 16) | ((tdim1 & 0xFFFFu) << 16)); // tensor_dim0 hi | tensor_dim1 lo
    g1[3] = (int)((tdim1 >> 16) | (tile0 << 16));             // tensor_dim1 hi | tile_dim0
    g1[4] = (int)(tile1 & 0xFFFFu);                           // tile_dim1 | tile_dim2=0
    g1[5] = (int)(unsigned)(stride0 & 0xFFFFFFFFu);           // tensor_dim0_stride lo32
    g1[6] = (int)(unsigned)((stride0 >> 32) & 0xFFFFu);       // stride hi16 | dim1_stride lo=0
    g1[7] = 0;
    v4i z4 = {0, 0, 0, 0};
#if defined(__clang_major__) && __clang_major__ >= 23
    v8i z8 = {0, 0, 0, 0, 0, 0, 0, 0};
    __builtin_amdgcn_tensor_load_to_lds(g0, g1, z4, z4, z8, 0);
#else
    __builtin_amdgcn_tensor_load_to_lds(g0, g1, z4, z4, 0);
#endif
}
#endif

#define KC    128            // K-chunk (floats)
#define BROWS 32             // B panel rows (output cols per block)
#define BPAD  132            // LDS row stride: 128 + TDM pad (4 dw per 128 dw)
#define CPAD  33             // CPE LDS row stride: 32 + 1

// ---------------- GEMM: Y[rows,dout) = act(X[rows,din] @ W[dout,din]^T + bias) ----------------
// 4 waves/block -> 64x32 tile; B panel TDM-staged to LDS (double buffered), A direct b128 loads.
__global__ __launch_bounds__(128)
void k_gemm(const float* __restrict__ X, int ldx,
            const float* __restrict__ W, const float* __restrict__ bias,
            float* __restrict__ Y, int ldy,
            int rows_s, const int* __restrict__ rows_d,
            int din, int dout, int act)
{
    __shared__ float Bs[2][BROWS * BPAD];
    int rows = nrows(rows_s, rows_d);
    if ((int)blockIdx.x * 64 >= rows) return;          // block-uniform
    int wave = threadIdx.x >> 5;
    int lane = threadIdx.x & 31, lhalf = lane >> 4, l16 = lane & 15;
    int tm = blockIdx.x * 64 + wave * 16, tn = blockIdx.y * 32;
    int arow = tm + l16;
    int arc  = arow < rows ? arow : rows - 1;          // clamp: garbage rows never stored
    const float* Xr = X + (size_t)arc * ldx;
    int bc0 = tn + l16, bc1 = tn + 16 + l16;
    int bcc0 = bc0 < dout ? bc0 : dout - 1;
    int bcc1 = bc1 < dout ? bc1 : dout - 1;
    v8f acc0 = {}, acc1 = {};

    if ((din & (KC - 1)) == 0) {                       // all large GEMMs here: din % 128 == 0
        int NC = din >> 7;
        const float* Wt = W + (size_t)tn * din;
#if HAVE_TDM
        if (wave == 0)
            tdm_load_2d(Wt, ldsoff(&Bs[0][0]), 0x7920000u,
                        (unsigned)din, (unsigned)(dout - tn), KC, BROWS, (unsigned long long)din);
        for (int cch = 0; cch < NC; ++cch) {
            if (wave == 0) __builtin_amdgcn_s_wait_tensorcnt(0);
            __syncthreads();                            // chunk cch visible; prev compute done
            if (wave == 0 && cch + 1 < NC)
                tdm_load_2d(Wt + (size_t)(cch + 1) * KC, ldsoff(&Bs[(cch + 1) & 1][0]), 0x7920000u,
                            (unsigned)(din - (cch + 1) * KC), (unsigned)(dout - tn),
                            KC, BROWS, (unsigned long long)din);
            const float* Bp = Bs[cch & 1];
            __builtin_prefetch(Xr + (size_t)(cch + 1) * KC, 0, 1);   // speculative, safe
#else
        for (int cch = 0; cch < NC; ++cch) {
            __syncthreads();
#pragma unroll
            for (int u = 0; u < 8; ++u) {               // cooperative copy fallback
                int e4 = threadIdx.x + u * 128;
                int rowi = e4 >> 5, c4 = e4 & 31;
                int wr = tn + rowi; wr = wr < dout ? wr : dout - 1;
                float4 q = *(const float4*)(W + (size_t)wr * din + (size_t)cch * KC + c4 * 4);
                *(float4*)&Bs[0][rowi * BPAD + c4 * 4] = q;
            }
            __syncthreads();
            const float* Bp = Bs[0];
#endif
#pragma unroll
            for (int kk = 0; kk < 4; ++kk) {
                int kb = kk * 32;
                const float4* pa0 = (const float4*)(Xr + (size_t)cch * KC + kb + lhalf * 16);
                const float4* pa1 = (const float4*)(Xr + (size_t)cch * KC + kb + 8 + lhalf * 16);
                float4 a00 = pa0[0], a01 = pa0[1], a10 = pa1[0], a11 = pa1[1];
                const float4* pb0 = (const float4*)&Bp[l16 * BPAD + kb + lhalf * 16];
                const float4* pb1 = (const float4*)&Bp[(16 + l16) * BPAD + kb + lhalf * 16];
                float4 q00 = pb0[0], q01 = pb0[1], q02 = pb0[2], q03 = pb0[3];
                float4 q10 = pb1[0], q11 = pb1[1], q12 = pb1[2], q13 = pb1[3];
                v16bf a, b0, b1;
                PK4(a, 0, a00)  PK4(a, 4, a01)  PK4(a, 8, a10)  PK4(a, 12, a11)
                PK4(b0, 0, q00) PK4(b0, 4, q01) PK4(b0, 8, q02) PK4(b0, 12, q03)
                PK4(b1, 0, q10) PK4(b1, 4, q11) PK4(b1, 8, q12) PK4(b1, 12, q13)
                acc0 = __builtin_amdgcn_wmma_f32_16x16x32_bf16(false, a, false, b0, (short)0, acc0, false, false);
                acc1 = __builtin_amdgcn_wmma_f32_16x16x32_bf16(false, a, false, b1, (short)0, acc1, false, false);
            }
        }
    } else {                                            // small-K path (din 6/8/16), per-wave
        const float* W0 = W + (size_t)bcc0 * din;
        const float* W1 = W + (size_t)bcc1 * din;
        int KT = (din + 31) >> 5;
        for (int kt = 0; kt < KT; ++kt) {
            int k0 = kt << 5;
            v16bf a, b0, b1;
#pragma unroll
            for (int i = 0; i < 8; ++i) {
                int k1 = k0 + lhalf * 16 + i, k2 = k1 + 8;
                float v1 = Xr[k1 < din ? k1 : 0];
                float v2 = Xr[k2 < din ? k2 : 0];
                a[i]     = tb(k1 < din ? v1 : 0.f);
                a[i + 8] = tb(k2 < din ? v2 : 0.f);
            }
#pragma unroll
            for (int i = 0; i < 16; ++i) {
                int k = k0 + lhalf * 16 + i;
                int kc = k < din ? k : 0;
                float w0 = W0[kc], w1 = W1[kc];
                b0[i] = tb(k < din ? w0 : 0.f);
                b1[i] = tb(k < din ? w1 : 0.f);
            }
            acc0 = __builtin_amdgcn_wmma_f32_16x16x32_bf16(false, a, false, b0, (short)0, acc0, false, false);
            acc1 = __builtin_amdgcn_wmma_f32_16x16x32_bf16(false, a, false, b1, (short)0, acc1, false, false);
        }
    }

    float bv0 = bias ? bias[bcc0] : 0.f;
    float bv1 = bias ? bias[bcc1] : 0.f;
    if (tm + 16 <= rows && tn + 32 <= dout) {
#pragma unroll
        for (int j = 0; j < 8; ++j) {
            int r = tm + j + lhalf * 8;
            Y[(size_t)r * ldy + bc0] = actf(acc0[j] + bv0, act);
            Y[(size_t)r * ldy + bc1] = actf(acc1[j] + bv1, act);
        }
    } else {
#pragma unroll
        for (int j = 0; j < 8; ++j) {
            int r = tm + j + lhalf * 8;
            if (r < rows && bc0 < dout) Y[(size_t)r * ldy + bc0] = actf(acc0[j] + bv0, act);
            if (r < rows && bc1 < dout) Y[(size_t)r * ldy + bc1] = actf(acc1[j] + bv1, act);
        }
    }
}

// ---------------- CPE gathered GEMM: y[n,d] = sum_k sum_c F[nbr[n,k],c] * CW[k,c,d] + cb --------
// CW is K-major [27c, d]; panel staged to LDS via TDM (128 K-rows x 32 cols, pad 1dw/32dw).
__global__ __launch_bounds__(128)
void k_gemm_cpe(const float* __restrict__ F, int cshift,
                const int* __restrict__ nbr,
                const float* __restrict__ CW, const float* __restrict__ cb,
                float* __restrict__ Y,
                int rows_s, const int* __restrict__ rows_d, int dout)
{
    __shared__ float Cs[2][KC * CPAD];
    int rows = nrows(rows_s, rows_d);
    if ((int)blockIdx.x * 64 >= rows) return;
    int c = 1 << cshift, din = 27 << cshift;
    int wave = threadIdx.x >> 5;
    int lane = threadIdx.x & 31, lhalf = lane >> 4, l16 = lane & 15;
    int tm = blockIdx.x * 64 + wave * 16, tn = blockIdx.y * 32;
    int arow = tm + l16;
    int arc  = arow < rows ? arow : rows - 1;
    const int* nb = nbr + (size_t)arc * 27;
    int bc0 = tn + l16, bc1 = tn + 16 + l16;            // dout multiple of 32
    v8f acc0 = {}, acc1 = {};
    int NC = din >> 7;
    const float* Wt = CW + tn;
#if HAVE_TDM
    if (wave == 0)
        tdm_load_2d(Wt, ldsoff(&Cs[0][0]), 0x1120000u,
                    (unsigned)(dout - tn), (unsigned)din, BROWS, KC, (unsigned long long)dout);
#endif
    for (int cch = 0; cch < NC; ++cch) {
#if HAVE_TDM
        if (wave == 0) __builtin_amdgcn_s_wait_tensorcnt(0);
        __syncthreads();
        if (wave == 0 && cch + 1 < NC)
            tdm_load_2d(Wt + (size_t)(cch + 1) * KC * dout, ldsoff(&Cs[(cch + 1) & 1][0]), 0x1120000u,
                        (unsigned)(dout - tn), (unsigned)(din - (cch + 1) * KC),
                        BROWS, KC, (unsigned long long)dout);
        const float* Bp = Cs[cch & 1];
#else
        __syncthreads();
#pragma unroll
        for (int u = 0; u < 8; ++u) {                   // cooperative copy fallback
            int e4 = threadIdx.x + u * 128;
            int rowi = e4 >> 3, c4 = e4 & 7;            // 128 rows x 8 float4
            float4 q = *(const float4*)(CW + (size_t)(cch * KC + rowi) * dout + tn + c4 * 4);
            *(float4*)&Cs[0][rowi * CPAD + c4 * 4] = q;
        }
        __syncthreads();
        const float* Bp = Cs[0];
#endif
#pragma unroll
        for (int kk = 0; kk < 4; ++kk) {
            int k0 = cch * KC + kk * 32;                // global K base
            int kl = kk * 32;                           // chunk-local K base
            v16bf a, b0, b1;
#pragma unroll
            for (int g = 0; g < 2; ++g) {
                int kkk = k0 + lhalf * 16 + g * 8;
                int kg = kkk >> cshift, cg = kkk & (c - 1);
                int idx = nb[kg];
                bool ok = (idx >= 0) && (idx < rows);
                const float4* p = (const float4*)(F + (size_t)(ok ? idx : 0) * c + cg);
                float4 q0 = p[0], q1 = p[1];
                q0.x = ok ? q0.x : 0.f; q0.y = ok ? q0.y : 0.f; q0.z = ok ? q0.z : 0.f; q0.w = ok ? q0.w : 0.f;
                q1.x = ok ? q1.x : 0.f; q1.y = ok ? q1.y : 0.f; q1.z = ok ? q1.z : 0.f; q1.w = ok ? q1.w : 0.f;
                PK4(a, g * 8, q0) PK4(a, g * 8 + 4, q1)
            }
#pragma unroll
            for (int i = 0; i < 16; ++i) {
                int kloc = kl + lhalf * 16 + i;
                b0[i] = tb(Bp[kloc * CPAD + l16]);
                b1[i] = tb(Bp[kloc * CPAD + 16 + l16]);
            }
            acc0 = __builtin_amdgcn_wmma_f32_16x16x32_bf16(false, a, false, b0, (short)0, acc0, false, false);
            acc1 = __builtin_amdgcn_wmma_f32_16x16x32_bf16(false, a, false, b1, (short)0, acc1, false, false);
        }
    }
    float bv0 = cb[bc0], bv1 = cb[bc1];
    if (tm + 16 <= rows) {
#pragma unroll
        for (int j = 0; j < 8; ++j) {
            int r = tm + j + lhalf * 8;
            Y[(size_t)r * dout + bc0] = acc0[j] + bv0;
            Y[(size_t)r * dout + bc1] = acc1[j] + bv1;
        }
    } else {
#pragma unroll
        for (int j = 0; j < 8; ++j) {
            int r = tm + j + lhalf * 8;
            if (r < rows) { Y[(size_t)r * dout + bc0] = acc0[j] + bv0;
                            Y[(size_t)r * dout + bc1] = acc1[j] + bv1; }
        }
    }
}

// ---------------- LayerNorm, d = 32*NV exactly ----------------
template <int NV>
__global__ void k_ln(const float* __restrict__ X, const float* __restrict__ R,
                     const float* __restrict__ w, const float* __restrict__ b,
                     float* __restrict__ Y, int ldy, int coloff,
                     int rows_s, const int* __restrict__ rows_d, int act)
{
    const int d = 32 * NV;
    int rows = nrows(rows_s, rows_d);
    int r = blockIdx.x;
    if (r >= rows) return;
    int lane = threadIdx.x;
    float xv[NV];
    float s = 0.f, ss = 0.f;
#pragma unroll
    for (int n = 0; n < NV; ++n) {
        int j = lane + 32 * n;
        float x = X[(size_t)r * d + j];
        if (R) x += R[(size_t)r * d + j];
        xv[n] = x; s += x; ss += x * x;
    }
    for (int m = 16; m; m >>= 1) { s += __shfl_xor(s, m); ss += __shfl_xor(ss, m); }
    float mean = s / d;
    float var  = ss / d - mean * mean;
    float rstd = rsqrtf(var + 1e-5f);
#pragma unroll
    for (int n = 0; n < NV; ++n) {
        int j = lane + 32 * n;
        float y = (xv[n] - mean) * rstd * w[j] + b[j];
        Y[(size_t)r * ldy + coloff + j] = actf(y, act);
    }
}

// ---------------- causal depthwise conv (DCONV=4) + silu ----------------
__global__ void k_conv(const float* __restrict__ XZ, int ldxz,
                       const float* __restrict__ w, const float* __restrict__ bias,
                       float* __restrict__ XC, int di, int rows_s, const int* __restrict__ rows_d)
{
    int rows = nrows(rows_s, rows_d);
    int t = blockIdx.x * blockDim.x + threadIdx.x;
    int ch = t % di, l = t / di;
    if (l >= rows) return;
    float acc = bias[ch];
#pragma unroll
    for (int k = 0; k < 4; ++k) {
        int ll = l - 3 + k;
        if (ll >= 0) acc += XZ[(size_t)ll * ldxz + ch] * w[ch * 4 + k];
    }
    XC[(size_t)l * di + ch] = siluf(acc);
}

// ---------------- selective scan ----------------
__global__ void k_scan(const float* __restrict__ dt, const float* __restrict__ dbc, int ldbc, int dtr,
                       const float* __restrict__ xc, const float* __restrict__ xz, int ldxz,
                       const float* __restrict__ alog, const float* __restrict__ Dp,
                       float* __restrict__ Y, int di, int rows_s, const int* __restrict__ rows_d)
{
    int rows = nrows(rows_s, rows_d);
    int t = blockIdx.x * blockDim.x + threadIdx.x;
    int ch = t >> 4, s = t & 15;
    if (ch >= di) return;
    float A   = -__expf(alog[ch * 16 + s]);
    float Dch = Dp[ch];
    float h = 0.f;
    for (int l = 0; l < rows; ++l) {
        float dtv = dt[(size_t)l * di + ch];
        float Bv  = dbc[(size_t)l * ldbc + dtr + s];
        float Cv  = dbc[(size_t)l * ldbc + dtr + 16 + s];
        float xcv = xc[(size_t)l * di + ch];
        h = __expf(dtv * A) * h + (dtv * xcv) * Bv;
        float p = h * Cv;
        p += __shfl_xor(p, 1); p += __shfl_xor(p, 2); p += __shfl_xor(p, 4); p += __shfl_xor(p, 8);
        if (s == 0) {
            float zv = xz[(size_t)l * ldxz + di + ch];
            Y[(size_t)l * di + ch] = (p + xcv * Dch) * siluf(zv);
        }
    }
}

// ---------------- scatter / gather rows ----------------
__global__ void k_scatter(const float* __restrict__ S, int lds,
                          const int* __restrict__ idxb, int idxrow,
                          float* __restrict__ D, int ldd, int coloff,
                          int rows_s, const int* __restrict__ rows_d, int cols)
{
    int rows = nrows(rows_s, rows_d);
    int i = blockIdx.x;
    if (i >= rows) return;
    int di_ = idxb ? idxb[(size_t)idxrow * rows + i] : i;
    for (int j = threadIdx.x; j < cols; j += blockDim.x)
        D[(size_t)di_ * ldd + coloff + j] = S[(size_t)i * lds + j];
}
__global__ void k_gather(const float* __restrict__ S, int lds,
                         const int* __restrict__ idx,
                         float* __restrict__ D, int ldd, int coloff,
                         int rows_s, const int* __restrict__ rows_d, int cols)
{
    int rows = nrows(rows_s, rows_d);
    int i = blockIdx.x;
    if (i >= rows) return;
    int si = idx[i];
    for (int j = threadIdx.x; j < cols; j += blockDim.x)
        D[(size_t)i * ldd + coloff + j] = S[(size_t)si * lds + j];
}

// ---------------- segment max ----------------
static __device__ __forceinline__ unsigned fenc(float f) {
    unsigned u = __builtin_bit_cast(unsigned, f);
    return (u & 0x80000000u) ? ~u : (u | 0x80000000u);
}
__global__ void k_segmax(const float* __restrict__ PF, const int* __restrict__ cl,
                         unsigned* __restrict__ P1e, int N, int cols)
{
    int t = blockIdx.x * blockDim.x + threadIdx.x;
    if (t >= N * cols) return;
    int n = t / cols, cc = t % cols;
    atomicMax(&P1e[(size_t)cl[n] * cols + cc], fenc(PF[t]));
}
__global__ void k_segdec(const unsigned* __restrict__ E, float* __restrict__ P, int n)
{
    int t = blockIdx.x * blockDim.x + threadIdx.x;
    if (t >= n) return;
    unsigned e = E[t];
    unsigned u = (e & 0x80000000u) ? (e ^ 0x80000000u) : ~e;
    P[t] = __builtin_bit_cast(float, u);
}

__global__ void k_cat6(const float* __restrict__ c3, const float* __restrict__ f3,
                       float* __restrict__ X6, int N)
{
    int t = blockIdx.x * blockDim.x + threadIdx.x;
    if (t >= N * 6) return;
    int n = t / 6, j = t % 6;
    X6[t] = j < 3 ? c3[n * 3 + j] : f3[n * 3 + j - 3];
}

// ======================= host side =======================
struct MambaP { const float *nw,*nb,*inp,*convw,*convb,*xproj,*dtw,*dtb,*alog,*Dp,*outp; };
struct MixP   { MambaP b; const float *fw,*fb; };
struct StageP {
    const float *cw,*cb,*cfw,*cfb,*clw,*clb;
    MixP mx[2];
    const float *fw1,*fb1,*flw,*flb,*fw2,*fb2;
};
struct Work { float *f,*f0,*p1,*hcpe,*hc,*seq,*xn,*xz,*xc,*dbc,*dt,*y,*mo; };

static inline int cdiv(int a, int b) { return (a + b - 1) / b; }

static void gemm(hipStream_t st, const float* X, int ldx, const float* W, const float* bias,
                 float* Y, int ldy, int Lmax, const int* rd, int din, int dout, int act)
{
    dim3 g(cdiv(Lmax, 64), cdiv(dout, 32));
    k_gemm<<<g, 128, 0, st>>>(X, ldx, W, bias, Y, ldy, Lmax, rd, din, dout, act);
}
static void ln(hipStream_t st, const float* X, const float* R, const float* w, const float* b,
               float* Y, int ldy, int coloff, int Lmax, const int* rd, int d, int act)
{
    if (d == 128) k_ln<4><<<Lmax, 32, 0, st>>>(X, R, w, b, Y, ldy, coloff, Lmax, rd, act);
    else          k_ln<8><<<Lmax, 32, 0, st>>>(X, R, w, b, Y, ldy, coloff, Lmax, rd, act);
}

static void run_stage(hipStream_t st, const StageP& P, int d, int Lmax, const int* rd,
                      const float* Fin, float* Fout,
                      const int* order, const int* inv, const int* nbr, Work& w)
{
    int di = 2 * d, dtr = d / 16, cshift = (d == 128) ? 7 : 8;
    dim3 gc(cdiv(Lmax, 64), cdiv(d, 32));
    k_gemm_cpe<<<gc, 128, 0, st>>>(Fin, cshift, nbr, P.cw, P.cb, w.seq, Lmax, rd, d);
    gemm(st, w.seq, d, P.cfw, P.cfb, w.xn, d, Lmax, rd, d, d, 0);
    ln(st, w.xn, Fin, P.clw, P.clb, w.hcpe, d, 0, Lmax, rd, d, 0);
    k_scatter<<<Lmax, 128, 0, st>>>(w.hcpe, d, nullptr, 0, w.hc, 3 * d, 0, Lmax, rd, d);
    for (int m = 0; m < 2; ++m) {
        const MixP& MX = P.mx[m];
        const MambaP& B = MX.b;
        k_scatter<<<Lmax, 128, 0, st>>>(w.hcpe, d, order, m, w.seq, d, 0, Lmax, rd, d);
        ln(st, w.seq, nullptr, B.nw, B.nb, w.xn, d, 0, Lmax, rd, d, 0);
        gemm(st, w.xn, d, B.inp, nullptr, w.xz, 2 * di, Lmax, rd, d, 2 * di, 0);
        k_conv<<<cdiv(Lmax * di, 256), 256, 0, st>>>(w.xz, 2 * di, B.convw, B.convb, w.xc, di, Lmax, rd);
        gemm(st, w.xc, di, B.xproj, nullptr, w.dbc, dtr + 32, Lmax, rd, di, dtr + 32, 0);
        gemm(st, w.dbc, dtr + 32, B.dtw, B.dtb, w.dt, di, Lmax, rd, dtr, di, 2);
        k_scan<<<di * 16 / 256, 256, 0, st>>>(w.dt, w.dbc, dtr + 32, dtr, w.xc, w.xz, 2 * di,
                                              B.alog, B.Dp, w.y, di, Lmax, rd);
        gemm(st, w.y, di, B.outp, nullptr, w.mo, d, Lmax, rd, di, d, 0);
        ln(st, w.mo, w.seq, MX.fw, MX.fb, w.xn, d, 0, Lmax, rd, d, 0);
        k_scatter<<<Lmax, 128, 0, st>>>(w.xn, d, inv, m, w.hc, 3 * d, (m + 1) * d, Lmax, rd, d);
    }
    gemm(st, w.hc, 3 * d, P.fw1, P.fb1, w.seq, d, Lmax, rd, 3 * d, d, 0);
    ln(st, w.seq, nullptr, P.flw, P.flb, w.xn, d, 0, Lmax, rd, d, 1);
    gemm(st, w.xn, d, P.fw2, P.fb2, Fout, d, Lmax, rd, d, d, 0);
}

extern "C" void kernel_launch(void* const* d_in, const int* in_sizes, int n_in,
                              void* d_out, int out_size, void* d_ws, size_t ws_size,
                              hipStream_t stream)
{
    (void)in_sizes; (void)n_in; (void)out_size; (void)ws_size;
    const int N = 8192, L = 8192;
    int i = 0;
    auto F = [&]() { return (const float*)d_in[i++]; };
    auto I = [&]() { return (const int*)d_in[i++]; };
    auto RS = [&]() {
        StageP P;
        P.cw = F(); P.cb = F(); P.cfw = F(); P.cfb = F(); P.clw = F(); P.clb = F();
        for (int m = 0; m < 2; ++m) {
            MambaP& B = P.mx[m].b;
            B.nw = F(); B.nb = F(); B.inp = F(); B.convw = F(); B.convb = F();
            B.xproj = F(); B.dtw = F(); B.dtb = F(); B.alog = F(); B.Dp = F(); B.outp = F();
            P.mx[m].fw = F(); P.mx[m].fb = F();
        }
        P.fw1 = F(); P.fb1 = F(); P.flw = F(); P.flb = F(); P.fw2 = F(); P.fb2 = F();
        return P;
    };

    const float* coord = F(); const float* feat = F();
    const float *ew1 = F(), *eb1 = F(), *ew2 = F(), *eb2 = F();
    StageP e0 = RS();
    const float *pw = F(), *pb = F(), *plw = F(), *plb = F();
    StageP e1 = RS();
    const float *upw = F(), *upb = F(), *uplw = F(), *uplb = F();
    const float *usw = F(), *usb = F(), *uslw = F(), *uslb = F();
    const float *ufw1 = F(), *ufb1 = F(), *uflw = F(), *uflb = F();
    const float *ufw2 = F(), *ufb2 = F();
    StageP d0 = RS();
    const int* Mptr = I();
    const int *order0 = I(), *inv0 = I(), *nbr0 = I(), *cluster = I();
    const int *order1 = I(), *inv1 = I(), *nbr1 = I();

    float* base = (float*)d_ws;
    size_t off = 0;
    auto alloc = [&](size_t n) { float* p = base + off; off += n; return p; };
    Work w;
    w.f    = alloc((size_t)L * 256);
    w.f0   = alloc((size_t)L * 128);
    w.p1   = alloc((size_t)L * 256);
    w.hcpe = alloc((size_t)L * 256);
    w.hc   = alloc((size_t)L * 768);
    w.seq  = alloc((size_t)L * 256);
    w.xn   = alloc((size_t)L * 256);
    w.xz   = alloc((size_t)L * 1024);
    w.xc   = alloc((size_t)L * 512);
    w.dbc  = alloc((size_t)L * 48);
    w.dt   = alloc((size_t)L * 512);
    w.y    = alloc((size_t)L * 512);
    w.mo   = alloc((size_t)L * 256);

    // encoder
    k_cat6<<<cdiv(N * 6, 256), 256, 0, stream>>>(coord, feat, w.dbc, N);
    gemm(stream, w.dbc, 6, ew1, eb1, w.seq, 256, N, nullptr, 6, 256, 1);
    gemm(stream, w.seq, 256, ew2, eb2, w.f, 128, N, nullptr, 256, 128, 0);

    // stage_e0 (d=128)
    run_stage(stream, e0, 128, N, nullptr, w.f, w.f0, order0, inv0, nbr0, w);

    // pool: GEMM -> segment max -> LN+gelu
    float* pf = w.xz;
    gemm(stream, w.f0, 128, pw, pb, pf, 256, N, nullptr, 128, 256, 0);
    unsigned* p1e = (unsigned*)w.xc;
    hipMemsetAsync(p1e, 0, (size_t)L * 256 * sizeof(unsigned), stream);
    k_segmax<<<cdiv(N * 256, 256), 256, 0, stream>>>(pf, cluster, p1e, N, 256);
    k_segdec<<<cdiv(L * 256, 256), 256, 0, stream>>>(p1e, w.p1, L * 256);
    ln(stream, w.p1, nullptr, plw, plb, w.f, 256, 0, L, Mptr, 256, 1);

    // stage_e1 (d=256, M rows dynamic)
    run_stage(stream, e1, 256, L, Mptr, w.f, w.p1, order1, inv1, nbr1, w);

    // unpool
    gemm(stream, w.f0, 128, usw, usb, w.seq, 128, N, nullptr, 128, 128, 0);
    ln(stream, w.seq, nullptr, uslw, uslb, w.xc, 256, 0, N, nullptr, 128, 1);        // fused[:, :128]
    gemm(stream, w.p1, 256, upw, upb, w.xn, 128, L, Mptr, 256, 128, 0);
    ln(stream, w.xn, nullptr, uplw, uplb, w.seq, 128, 0, L, Mptr, 128, 1);           // prj [M,128]
    k_gather<<<N, 128, 0, stream>>>(w.seq, 128, cluster, w.xc, 256, 128, N, nullptr, 128);
    gemm(stream, w.xc, 256, ufw1, ufb1, w.xn, 128, N, nullptr, 256, 128, 0);
    ln(stream, w.xn, nullptr, uflw, uflb, w.seq, 128, 0, N, nullptr, 128, 1);
    gemm(stream, w.seq, 128, ufw2, ufb2, w.f, 128, N, nullptr, 128, 128, 0);

    // stage_d0 (d=128) -> final output
    run_stage(stream, d0, 128, N, nullptr, w.f, (float*)d_out, order0, inv0, nbr0, w);
}
